// GATLayer_73538430042454
// MI455X (gfx1250) — matrix operand order repaired and checked by reading, not compiled, not run on hardware
//
#include <hip/hip_runtime.h>

typedef __attribute__((ext_vector_type(16))) _Float16 v16h;
typedef __attribute__((ext_vector_type(8)))  _Float16 v8h;
typedef __attribute__((ext_vector_type(8)))  float    v8f;

#define GH 4
#define GB 4
#define GN 2048
#define GF 128
#define GU 64
#define PSTRIDE 40    // halves per u-row of a staged projT chunk (32 data + 8 pad)
#define WSTRIDE 136   // halves per u-row of staged W (128 data + 8 pad)

__device__ __forceinline__ v8f wmma16(v16h a, v16h b, v8f c) {
  return __builtin_amdgcn_wmma_f32_16x16x32_f16(false, a, false, b, (short)0, c, false, false);
}

// Async global->LDS 16B copy (CDNA5 GLOBAL_LOAD_ASYNC_TO_LDS_B128, ASYNCcnt).
// LDS byte address = low 32 bits of the generic pointer (ISA aperture rule).
__device__ __forceinline__ void async_copy_b128(const void* g, void* l) {
  unsigned lds = (unsigned)(size_t)l;
  unsigned long long ga = (unsigned long long)(size_t)g;
  asm volatile("global_load_async_to_lds_b128 %0, %1, off" :: "v"(lds), "v"(ga) : "memory");
}
__device__ __forceinline__ void wait_async0() {
  asm volatile("s_wait_asynccnt 0" ::: "memory");
}

// ---------------------------------------------------------------------------
// Kernel 1: proj[h,b,n,u] = relu(x[b,n,:] @ W[h,:,:] + bias[h,:]), stored f16
// transposed as projT[h][b][u][n]. W[h] staged once per block in LDS (f16).
// One wave = 16 rows x 64 u. grid (N/128, B, H), block 256 (8 waves).
// ---------------------------------------------------------------------------
__global__ __launch_bounds__(256) void gat_proj(const float* __restrict__ x,
                                                const float* __restrict__ W,
                                                const float* __restrict__ bias,
                                                _Float16* __restrict__ projT) {
  __shared__ __align__(16) _Float16 wS[GU * WSTRIDE];   // ~17 KB
  const int h = blockIdx.z, b = blockIdx.y;
  const int wv = threadIdx.x >> 5, lane = threadIdx.x & 31;
  const int g = lane >> 4, m = lane & 15;
  const int rowbase = blockIdx.x * 128 + wv * 16;
  const int row = rowbase + m;

  // stage W[h] transposed (u-major), f32 -> f16; u fastest for coalescing
  for (int idx = threadIdx.x; idx < GU * GF; idx += 256) {
    const int u = idx & (GU - 1), f = idx >> 6;
    wS[u * WSTRIDE + f] = (_Float16)W[((size_t)h * GF + f) * GU + u];
  }
  __syncthreads();

  v8f acc[4] = {};
#pragma unroll
  for (int fc = 0; fc < 4; ++fc) {
    // A operand: x[b,row, fc*32 .. +31], vectorized f32 loads -> f16 A layout
    const float4* xv = (const float4*)(x + ((size_t)b * GN + row) * GF + fc * 32);
    const float4 p0 = xv[g * 2 + 0];
    const float4 p1 = xv[g * 2 + 1];
    const float4 p2 = xv[g * 2 + 4];
    const float4 p3 = xv[g * 2 + 5];
    v16h a;
    a[0] = (_Float16)p0.x;  a[1] = (_Float16)p0.y;  a[2] = (_Float16)p0.z;  a[3] = (_Float16)p0.w;
    a[4] = (_Float16)p1.x;  a[5] = (_Float16)p1.y;  a[6] = (_Float16)p1.z;  a[7] = (_Float16)p1.w;
    a[8] = (_Float16)p2.x;  a[9] = (_Float16)p2.y;  a[10] = (_Float16)p2.z; a[11] = (_Float16)p2.w;
    a[12] = (_Float16)p3.x; a[13] = (_Float16)p3.y; a[14] = (_Float16)p3.z; a[15] = (_Float16)p3.w;
#pragma unroll
    for (int uc = 0; uc < 4; ++uc) {
      // B operand from LDS: K=8g+t (t<8), K=16+8g+(t-8) (t>=8), f = fc*32 + K
      const int off = (uc * 16 + m) * WSTRIDE + fc * 32 + g * 8;
      const v8h lo = *(const v8h*)&wS[off];
      const v8h hi = *(const v8h*)&wS[off + 16];
      v16h bm;
#pragma unroll
      for (int t = 0; t < 8; ++t) { bm[t] = lo[t]; bm[8 + t] = hi[t]; }
      acc[uc] = wmma16(a, bm, acc[uc]);
    }
  }
  // epilogue: bias + relu, store transposed f16
#pragma unroll
  for (int uc = 0; uc < 4; ++uc) {
    const int u = uc * 16 + m;
    const float bb = bias[h * GU + u];
    _Float16* pc = projT + ((size_t)(h * GB + b) * GU + u) * GN;
#pragma unroll
    for (int r = 0; r < 8; ++r) {
      const int i = rowbase + r + 8 * g;
      float v = acc[uc][r] + bb;
      pc[i] = (_Float16)(v > 0.f ? v : 0.f);
    }
  }
}

// ---------------------------------------------------------------------------
// Kernel 2: s[h,b,n] = sum_u projT[h,b,u,n]*a_w[h,u] + a_b[h]
// ---------------------------------------------------------------------------
__global__ __launch_bounds__(256) void gat_score(const _Float16* __restrict__ projT,
                                                 const float* __restrict__ a_w,
                                                 const float* __restrict__ a_b,
                                                 float* __restrict__ sW) {
  const int idx = blockIdx.x * 256 + threadIdx.x;   // H*B*N = 32768
  const int n = idx & (GN - 1);
  const int b = (idx >> 11) & (GB - 1);
  const int h = idx >> 13;
  const _Float16* p = projT + ((size_t)(h * GB + b) * GU) * GN + n;
  float s = a_b[h];
#pragma unroll 8
  for (int u = 0; u < GU; ++u) s += (float)p[(size_t)u * GN] * a_w[h * GU + u];
  sW[(h * GB + b) * GN + n] = s;
}

// ---------------------------------------------------------------------------
// Kernel 3: per row i, all heads: masked softmax stats (m_i, 1/sum) using the
// rank-1 structure of logits, plus adjacency row -> bitmask compression.
// One wave per row. grid (N/8, B), block 256 (8 waves).
// ---------------------------------------------------------------------------
__global__ __launch_bounds__(256) void gat_stats(const int* __restrict__ adj,
                                                 const float* __restrict__ sW,
                                                 float* __restrict__ mW,
                                                 float* __restrict__ rW,
                                                 unsigned* __restrict__ maskW) {
  __shared__ float sS4[GH * GN];   // 32 KB
  const int b = blockIdx.y;
  const int wv = threadIdx.x >> 5, lane = threadIdx.x & 31;
  const int i = blockIdx.x * 8 + wv;

  for (int idx = threadIdx.x; idx < GH * GN; idx += 256) {
    const int h = idx >> 11, j = idx & (GN - 1);
    sS4[idx] = sW[(h * GB + b) * GN + j];
  }
  __syncthreads();

  float si[GH];
#pragma unroll
  for (int h = 0; h < GH; ++h) si[h] = sS4[h * GN + i];

  const int* arow = adj + ((size_t)b * GN + i) * GN;
  unsigned* mrow = maskW + ((size_t)b * GN + i) * (GN / 32);

  float mx[GH], mn[GH];
#pragma unroll
  for (int h = 0; h < GH; ++h) { mx[h] = -3.0e38f; mn[h] = 3.0e38f; }

  // pass 1: bitmask + masked max/min of s_j
  for (int jc = 0; jc < GN / 32; ++jc) {
    const int j = jc * 32 + lane;
    const int a = arow[j];
    unsigned word = (a > 0 ? 1u : 0u) << lane;
#pragma unroll
    for (int off = 16; off > 0; off >>= 1) word |= __shfl_xor(word, off, 32);
    if (lane == 0) mrow[jc] = word;
    if (a > 0) {
#pragma unroll
      for (int h = 0; h < GH; ++h) {
        const float v = sS4[h * GN + j];
        mx[h] = fmaxf(mx[h], v);
        mn[h] = fminf(mn[h], v);
      }
    }
  }
#pragma unroll
  for (int off = 16; off > 0; off >>= 1) {
#pragma unroll
    for (int h = 0; h < GH; ++h) {
      mx[h] = fmaxf(mx[h], __shfl_xor(mx[h], off, 32));
      mn[h] = fminf(mn[h], __shfl_xor(mn[h], off, 32));
    }
  }
  float m[GH];
#pragma unroll
  for (int h = 0; h < GH; ++h) m[h] = (si[h] >= 0.f) ? si[h] * mx[h] : si[h] * mn[h];

  // pass 2: sum of exp over masked j
  float sum[GH] = {};
  for (int jc = 0; jc < GN / 32; ++jc) {
    const int j = jc * 32 + lane;
    if (arow[j] > 0) {
#pragma unroll
      for (int h = 0; h < GH; ++h)
        sum[h] += __expf(si[h] * sS4[h * GN + j] - m[h]);
    }
  }
#pragma unroll
  for (int off = 16; off > 0; off >>= 1) {
#pragma unroll
    for (int h = 0; h < GH; ++h) sum[h] += __shfl_xor(sum[h], off, 32);
  }
  if (lane == 0) {
#pragma unroll
    for (int h = 0; h < GH; ++h) {
      mW[(h * GB + b) * GN + i] = m[h];
      rW[(h * GB + b) * GN + i] = (sum[h] > 1e-30f) ? (1.f / sum[h]) : 0.f;
    }
  }
}

// ---------------------------------------------------------------------------
// Kernel 4: out[b,i,u] = mean_h sum_j P[h,b,i,j] * proj[h,b,j,u]
// Flash-style, P never materialized. One wave = 16-row tile; block = 8 waves
// (128 rows). projT K-chunks double-buffered in LDS via async global->LDS
// copies (ASYNCcnt); adjacency bitmasks staged once per block. f32
// accumulators kept across all 4 heads -> head mean at store.
// grid (N/128, B), block 256.
// ---------------------------------------------------------------------------
__global__ __launch_bounds__(256) void gat_agg(const _Float16* __restrict__ projT,
                                               const float* __restrict__ sW,
                                               const float* __restrict__ mW,
                                               const float* __restrict__ rW,
                                               const unsigned* __restrict__ maskW,
                                               float* __restrict__ out) {
  __shared__ float sS[GN];                                    // 8 KB
  __shared__ __align__(16) _Float16 projS[2][GU * PSTRIDE];   // 10 KB (double buffer)
  __shared__ unsigned maskS[128 * (GN / 32)];                 // 32 KB
  const int b = blockIdx.y;
  const int wv = threadIdx.x >> 5, lane = threadIdx.x & 31;
  const int g = lane >> 4, m = lane & 15;
  const int rowblock = blockIdx.x * 128;
  const int rowbase = rowblock + wv * 16;
  const int row = rowbase + m;

  // stage adjacency bitmasks for this block's 128 rows (head-invariant)
  {
    const unsigned* msrc = maskW + ((size_t)b * GN + rowblock) * (GN / 32);
    for (int idx = threadIdx.x; idx < 128 * (GN / 32); idx += 256)
      maskS[idx] = msrc[idx];
  }
  const unsigned* mrow = maskS + (wv * 16 + m) * (GN / 32);

  // per-thread staging coords: thread copies 16B of projT chunk (u = tid/4)
  const int su = threadIdx.x >> 2, sq = threadIdx.x & 3;

  v8f acc[4] = {};

  for (int h = 0; h < GH; ++h) {
    const int hb = h * GB + b;
    __syncthreads();   // prior head done with sS / buffers
    const _Float16* srcrow = projT + ((size_t)hb * GU + su) * GN + sq * 8;
    // prologue: async-stage chunk 0 into buffer 0, overlap with sS staging
    async_copy_b128(srcrow, &projS[0][su * PSTRIDE + sq * 8]);
    for (int idx = threadIdx.x; idx < GN; idx += 256)
      sS[idx] = sW[(size_t)hb * GN + idx];
    wait_async0();
    __syncthreads();

    const float si = sS[row];
    const float mi = mW[(size_t)hb * GN + row];
    const float ri = rW[(size_t)hb * GN + row];

    for (int jc = 0; jc < GN / 32; ++jc) {
      const int jb = jc * 32;
      const int cur = jc & 1;
      if (jc + 1 < GN / 32)   // prefetch next chunk into the other buffer
        async_copy_b128(srcrow + jb + 32, &projS[cur ^ 1][su * PSTRIDE + sq * 8]);

      // A operand: normalized softmax weights P, exact 16x32 f16 layout
      const unsigned mw = mrow[jc];
      v16h a;
#pragma unroll
      for (int t = 0; t < 8; ++t) {
        const int k0 = g * 8 + t;
        const int k1 = 16 + g * 8 + t;
        a[t]     = ((mw >> k0) & 1u) ? (_Float16)(__expf(si * sS[jb + k0] - mi) * ri)
                                     : (_Float16)0.f;
        a[8 + t] = ((mw >> k1) & 1u) ? (_Float16)(__expf(si * sS[jb + k1] - mi) * ri)
                                     : (_Float16)0.f;
      }
      const _Float16* pb = projS[cur];
#pragma unroll
      for (int uc = 0; uc < 4; ++uc) {
        const int off = (uc * 16 + m) * PSTRIDE + g * 8;
        const v8h lo = *(const v8h*)&pb[off];
        const v8h hi = *(const v8h*)&pb[off + 16];
        v16h bm;
#pragma unroll
        for (int t = 0; t < 8; ++t) { bm[t] = lo[t]; bm[8 + t] = hi[t]; }
        acc[uc] = wmma16(a, bm, acc[uc]);
      }

      if (jc + 1 < GN / 32) {   // next chunk arrived everywhere before reuse
        wait_async0();
        __syncthreads();
      }
    }
  }

  // head mean + store (fully overwrites d_out, no init needed)
#pragma unroll
  for (int uc = 0; uc < 4; ++uc) {
    const int u = uc * 16 + m;
#pragma unroll
    for (int r = 0; r < 8; ++r) {
      const int i = rowbase + r + 8 * g;
      out[((size_t)b * GN + i) * GU + u] = acc[uc][r] * 0.25f;
    }
  }
}

// ---------------------------------------------------------------------------
extern "C" void kernel_launch(void* const* d_in, const int* in_sizes, int n_in,
                              void* d_out, int out_size, void* d_ws, size_t ws_size,
                              hipStream_t stream) {
  const float* x    = (const float*)d_in[0];
  const int*   adj  = (const int*)d_in[1];
  const float* W    = (const float*)d_in[2];
  const float* bias = (const float*)d_in[3];
  const float* a_w  = (const float*)d_in[4];
  const float* a_b  = (const float*)d_in[5];

  char* ws = (char*)d_ws;
  _Float16* projT = (_Float16*)ws;                                  // 4 MB
  float*    sW    = (float*)(ws + (size_t)GH * GB * GU * GN * 2);   // 128 KB
  float*    mW    = sW + GH * GB * GN;                              // 128 KB
  float*    rW    = mW + GH * GB * GN;                              // 128 KB
  unsigned* maskW = (unsigned*)(rW + GH * GB * GN);                 // 2 MB

  gat_proj <<<dim3(GN / 128, GB, GH), 256, 0, stream>>>(x, W, bias, projT);
  gat_score<<<(GH * GB * GN) / 256, 256, 0, stream>>>(projT, a_w, a_b, sW);
  gat_stats<<<dim3(GN / 8, GB), 256, 0, stream>>>(adj, sW, mW, rW, maskW);
  gat_agg  <<<dim3(GN / 128, GB), 256, 0, stream>>>(projT, sW, mW, rW, maskW,
                                                    (float*)d_out);
}
